// future_STGAT_decoder_9165460210138
// MI455X (gfx1250) — matrix operand order repaired and checked by reading, not compiled, not run on hardware
//
#include <hip/hip_runtime.h>
#include <hip/hip_bf16.h>

#define H        1024
#define FOUR_H   4096
#define NAG      4096
#define FUT      12
#define OBS      8

// GEMM block tile
#define BM       128
#define BN       64
#define BK       32
#define A_STRIDE 80   // 64B of K data + 16B pad (bank-conflict break)
#define B_STRIDE 80
#define A_BYTES  (BM * A_STRIDE)      // 10240
#define B_BYTES  (BN * B_STRIDE)      //  5120
#define BUF_BYTES (A_BYTES + B_BYTES) // 15360

typedef __bf16 bf16_t;
typedef __attribute__((ext_vector_type(16))) bf16_t v16bf;
typedef __attribute__((ext_vector_type(8)))  bf16_t v8bf;
typedef __attribute__((ext_vector_type(8)))  float  v8f;

// Exact vector type the async-LDS builtin expects (per hipcc diagnostic):
// 'int __attribute__((vector_size(16)))'
typedef int v4i __attribute__((vector_size(16)));
typedef __attribute__((address_space(1))) v4i* gv4i_p;  // global
typedef __attribute__((address_space(3))) v4i* lv4i_p;  // LDS

union AFrag {
    v16bf v;
    struct { v8bf lo; v8bf hi; } part;
};

#if __has_builtin(__builtin_amdgcn_global_load_async_to_lds_b128)
#define HAVE_ASYNC_LDS 1
#else
#define HAVE_ASYNC_LDS 0
#endif

// 16-byte global -> LDS copy; async (ASYNCcnt) when available.
__device__ __forceinline__ void cp16_g2l(const void* g, void* l) {
#if HAVE_ASYNC_LDS
    __builtin_amdgcn_global_load_async_to_lds_b128(
        (gv4i_p)g, (lv4i_p)l, 0, 0);
#else
    *(uint4*)l = *(const uint4*)g;
#endif
}

__device__ __forceinline__ void wait_async_copies() {
#if HAVE_ASYNC_LDS && __has_builtin(__builtin_amdgcn_s_wait_asynccnt)
    __builtin_amdgcn_s_wait_asynccnt(0);
#endif
}

// ---------------------------------------------------------------------------
// Kernel 0: convert w_hh (4H x H) and h0 (NAG x H) fp32 -> bf16, and copy
// out0 = obs_traj_rel[OBS-1] into out_prev. Runs once per launch.
// ---------------------------------------------------------------------------
__global__ __launch_bounds__(256)
void cvt_init_kernel(const float* __restrict__ w_hh,
                     const float* __restrict__ h0,
                     const float* __restrict__ obs,
                     bf16_t* __restrict__ w_bf,
                     bf16_t* __restrict__ h_bf,
                     float* __restrict__ out_prev)
{
    const size_t stride = (size_t)gridDim.x * blockDim.x;
    const size_t tid    = (size_t)blockIdx.x * blockDim.x + threadIdx.x;

    const size_t nW = (size_t)FOUR_H * H;
    for (size_t k = tid; k < nW; k += stride)
        w_bf[k] = (bf16_t)w_hh[k];

    const size_t nH = (size_t)NAG * H;
    for (size_t k = tid; k < nH; k += stride)
        h_bf[k] = (bf16_t)h0[k];

    if (tid < (size_t)NAG * 2)
        out_prev[tid] = obs[(size_t)(OBS - 1) * NAG * 2 + tid];
}

// ---------------------------------------------------------------------------
// Stage one BMxBK A-slice + BNxBK B-slice into an LDS buffer.
// 256 threads: A = 512 16B chunks (2/thread), B = 256 16B chunks (1/thread).
// LDS row layout: 64B of K data at rowStride 80B.
// ---------------------------------------------------------------------------
__device__ __forceinline__ void stage_tile(char* __restrict__ buf,
                                           const bf16_t* __restrict__ hbf,
                                           const bf16_t* __restrict__ wbf,
                                           int m0, int n0, int k0, int tid)
{
    char* Ab = buf;
    char* Bb = buf + A_BYTES;

    // A chunks: chunk c -> row c>>2, k-offset (c&3)*8 elements
    {
        int c   = tid;                       // 0..255
        int row = c >> 2, ko = (c & 3) * 8;
        cp16_g2l(hbf + (size_t)(m0 + row) * H + k0 + ko,
                 Ab + row * A_STRIDE + (c & 3) * 16);
        c   = tid + 256;                     // 256..511
        row = c >> 2; ko = (c & 3) * 8;
        cp16_g2l(hbf + (size_t)(m0 + row) * H + k0 + ko,
                 Ab + row * A_STRIDE + (c & 3) * 16);
    }
    // B chunks: chunk c -> w_hh row (column of B) c>>2
    {
        int c   = tid;
        int row = c >> 2, ko = (c & 3) * 8;
        cp16_g2l(wbf + (size_t)(n0 + row) * H + k0 + ko,
                 Bb + row * B_STRIDE + (c & 3) * 16);
    }
}

// ---------------------------------------------------------------------------
// Kernel 1: gates = h @ w_hh^T (bf16 WMMA, fp32 acc) + out@w_ih^T + biases.
// Block tile BMxBN staged through double-buffered LDS via async copies;
// 8 waves, wave w computes rows [w*16, w*16+16) x all BN columns.
// ---------------------------------------------------------------------------
__global__ __launch_bounds__(256)
void gates_gemm_kernel(const bf16_t* __restrict__ hbf,      // (NAG, H)
                       const bf16_t* __restrict__ wbf,      // (4H, H) row-major
                       const float*  __restrict__ w_ih,     // (4H, 2)
                       const float*  __restrict__ b_ih,     // (4H)
                       const float*  __restrict__ b_hh,     // (4H)
                       const float*  __restrict__ out_prev, // (NAG, 2)
                       float*        __restrict__ gates)    // (NAG, 4H)
{
    __shared__ __align__(16) char smem[2][BUF_BYTES];

    const int tid  = threadIdx.x;
    const int lane = tid & 31;
    const int wave = tid >> 5;

    const int bm = blockIdx.x & 31;   // 32 m-blocks
    const int bn = blockIdx.x >> 5;   // 64 n-blocks
    const int m0 = bm * BM;
    const int n0 = bn * BN;

    const int  l15    = lane & 15;
    const bool hiHalf = lane >= 16;

    // LDS fragment offsets (bytes)
    const int aRow = wave * 16 + l15;
    const int aOff = aRow * A_STRIDE + (hiHalf ? 32 : 0);   // lo chunk; hi at +16
    const int bOff = l15 * B_STRIDE + (hiHalf ? 32 : 0);    // per nt add 16*B_STRIDE

    v8f acc0 = {}, acc1 = {}, acc2 = {}, acc3 = {};

    stage_tile(smem[0], hbf, wbf, m0, n0, 0, tid);

    const int KSTEPS = H / BK;   // 32
    for (int kt = 0; kt < KSTEPS; ++kt) {
        wait_async_copies();     // this wave's copies into smem[kt&1] done
        __syncthreads();         // whole tile resident; prev compute finished

        if (kt + 1 < KSTEPS)
            stage_tile(smem[(kt + 1) & 1], hbf, wbf, m0, n0, (kt + 1) * BK, tid);

        const char* Ab = smem[kt & 1];
        const char* Bb = smem[kt & 1] + A_BYTES;

        // A fragment: lanes 0-15 K 0..15; lanes 16-31 K 16..31
        AFrag a;
        a.part.lo = *(const v8bf*)(Ab + aOff);
        a.part.hi = *(const v8bf*)(Ab + aOff + 16);

        // B fragments: lanes 0-15 K 0..15, lanes 16-31 K 16..31, col = l15
        v16bf b0 = *(const v16bf*)(Bb + bOff);
        v16bf b1 = *(const v16bf*)(Bb + bOff + 16 * B_STRIDE);
        v16bf b2 = *(const v16bf*)(Bb + bOff + 32 * B_STRIDE);
        v16bf b3 = *(const v16bf*)(Bb + bOff + 48 * B_STRIDE);

        acc0 = __builtin_amdgcn_wmma_f32_16x16x32_bf16(false, a.v, false, b0,
                                                       (short)0, acc0, false, false);
        acc1 = __builtin_amdgcn_wmma_f32_16x16x32_bf16(false, a.v, false, b1,
                                                       (short)0, acc1, false, false);
        acc2 = __builtin_amdgcn_wmma_f32_16x16x32_bf16(false, a.v, false, b2,
                                                       (short)0, acc2, false, false);
        acc3 = __builtin_amdgcn_wmma_f32_16x16x32_bf16(false, a.v, false, b3,
                                                       (short)0, acc3, false, false);
    }

    // Epilogue. C/D layout: lane l holds column n = l&15; VGPR v holds row
    // m = v + (l<16 ? 0 : 8) within the 16-row wave tile.
    const int mbase = m0 + wave * 16 + (hiHalf ? 8 : 0);
    float op0[8], op1[8];
#pragma unroll
    for (int v = 0; v < 8; ++v) {
        op0[v] = out_prev[(mbase + v) * 2 + 0];
        op1[v] = out_prev[(mbase + v) * 2 + 1];
    }

    v8f accs[4] = { acc0, acc1, acc2, acc3 };
#pragma unroll
    for (int nt = 0; nt < 4; ++nt) {
        const int   col  = n0 + nt * 16 + l15;
        const float bias = b_ih[col] + b_hh[col];
        const float wi0  = w_ih[col * 2 + 0];
        const float wi1  = w_ih[col * 2 + 1];
#pragma unroll
        for (int v = 0; v < 8; ++v) {
            const float val = accs[nt][v] + bias + op0[v] * wi0 + op1[v] * wi1;
            gates[(size_t)(mbase + v) * FOUR_H + col] = val;
        }
    }
}

// ---------------------------------------------------------------------------
// Kernel 2: per-agent LSTM cell + fused output projection.
// ---------------------------------------------------------------------------
__global__ __launch_bounds__(256)
void lstm_cell_kernel(const float* __restrict__ gates,   // (NAG, 4H)
                      float*       __restrict__ c_state, // (NAG, H)
                      bf16_t*      __restrict__ h_bf,    // (NAG, H) next A matrix
                      const float* __restrict__ w_out,   // (2, H)
                      const float* __restrict__ b_out,   // (2)
                      float*       __restrict__ out_prev,// (NAG, 2)
                      float*       __restrict__ pred_t,  // (NAG, 2) slice of d_out
                      int first_step)
{
    __shared__ float s0[256];
    __shared__ float s1[256];

    const int    agent = blockIdx.x;
    const float* g     = gates + (size_t)agent * FOUR_H;
    float*       crow  = c_state + (size_t)agent * H;
    bf16_t*      hrow  = h_bf + (size_t)agent * H;

    float acc0 = 0.f, acc1 = 0.f;
    for (int j = threadIdx.x; j < H; j += 256) {
        const float gi = g[j];
        const float gf = g[H + j];
        const float gg = g[2 * H + j];
        const float go = g[3 * H + j];

        const float si = 1.f / (1.f + __expf(-gi));
        const float sf = 1.f / (1.f + __expf(-gf));
        const float so = 1.f / (1.f + __expf(-go));
        const float tg = tanhf(gg);

        const float cp = first_step ? 0.f : crow[j];
        const float c  = sf * cp + si * tg;
        const float h  = so * tanhf(c);

        crow[j] = c;
        hrow[j] = (bf16_t)h;

        acc0 += h * w_out[j];
        acc1 += h * w_out[H + j];
    }

    s0[threadIdx.x] = acc0;
    s1[threadIdx.x] = acc1;
    __syncthreads();
    for (int off = 128; off > 0; off >>= 1) {
        if (threadIdx.x < off) {
            s0[threadIdx.x] += s0[threadIdx.x + off];
            s1[threadIdx.x] += s1[threadIdx.x + off];
        }
        __syncthreads();
    }

    if (threadIdx.x == 0) {
        const float o0 = s0[0] + b_out[0];
        const float o1 = s1[0] + b_out[1];
        out_prev[agent * 2 + 0] = o0;
        out_prev[agent * 2 + 1] = o1;
        pred_t[agent * 2 + 0] = o0;
        pred_t[agent * 2 + 1] = o1;
    }
}

// ---------------------------------------------------------------------------
// Host side: carve workspace, convert weights once, then 12 x (GEMM + cell).
// ---------------------------------------------------------------------------
extern "C" void kernel_launch(void* const* d_in, const int* in_sizes, int n_in,
                              void* d_out, int out_size, void* d_ws, size_t ws_size,
                              hipStream_t stream) {
    (void)in_sizes; (void)n_in; (void)out_size; (void)ws_size;

    const float* obs   = (const float*)d_in[0];  // (8, 4096, 2)
    const float* h0    = (const float*)d_in[1];  // (4096, 1024)
    const float* w_ih  = (const float*)d_in[2];  // (4096, 2)
    const float* w_hh  = (const float*)d_in[3];  // (4096, 1024)
    const float* b_ih  = (const float*)d_in[4];  // (4096)
    const float* b_hh  = (const float*)d_in[5];  // (4096)
    const float* w_out = (const float*)d_in[6];  // (2, 1024)
    const float* b_out = (const float*)d_in[7];  // (2)
    float*       out   = (float*)d_out;          // (12, 4096, 2)

    char* ws = (char*)d_ws;
    float*  gates    = (float*)(ws);                           // 64 MiB
    float*  c_state  = (float*)(ws + ((size_t)64 << 20));      // 16 MiB
    bf16_t* h_bf     = (bf16_t*)(ws + ((size_t)80 << 20));     //  8 MiB
    bf16_t* w_bf     = (bf16_t*)(ws + ((size_t)88 << 20));     //  8 MiB
    float*  out_prev = (float*)(ws + ((size_t)96 << 20));      // 32 KiB

    cvt_init_kernel<<<2048, 256, 0, stream>>>(w_hh, h0, obs, w_bf, h_bf, out_prev);

    // (NAG/BM)=32 m-blocks * (4H/BN)=64 n-blocks = 2048 blocks
    const int gemm_blocks = (NAG / BM) * (FOUR_H / BN);

    for (int t = 0; t < FUT; ++t) {
        gates_gemm_kernel<<<gemm_blocks, 256, 0, stream>>>(
            h_bf, w_bf, w_ih, b_ih, b_hh, out_prev, gates);
        lstm_cell_kernel<<<NAG, 256, 0, stream>>>(
            gates, c_state, h_bf, w_out, b_out, out_prev,
            out + (size_t)t * NAG * 2, t == 0 ? 1 : 0);
    }
}
